// ModelMAAR_58798102282341
// MI455X (gfx1250) — matrix operand order repaired and checked by reading, not compiled, not run on hardware
//
#include <hip/hip_runtime.h>
#include <math.h>

// ---------------------------------------------------------------------------
// MI455X (gfx1250, wave32) implementation.
// All heavy math routed through v_wmma_f32_16x16x32_bf16 (f32 accumulate).
// Convs are implicit-im2col GEMMs; attention is per-(l,head) 32x32 WMMA.
// All WMMA fragment traffic is b128-vectorized.
// ---------------------------------------------------------------------------

typedef __bf16 bf16_t;
typedef __attribute__((ext_vector_type(16))) __bf16 v16bf;
typedef __attribute__((ext_vector_type(8)))  float  v8f;

union FragU {
  v16bf v;
  uint4 q[2];
};
union Pack8 {
  uint4  q;
  bf16_t e[8];
};

static __device__ __forceinline__ float sigmoidf_(float x){
  return 1.0f / (1.0f + __expf(-x));
}

// ------------------------------ tiny kernels -------------------------------

__global__ void cvt_f32_bf16(const float* __restrict__ src, bf16_t* __restrict__ dst, int n){
  int i = blockIdx.x * blockDim.x + threadIdx.x;
  if (i < n) dst[i] = (bf16_t)src[i];
}

// out[b][c][l] = emb[idx[b][l]][c]   (bf16, conv input layout)
__global__ void embed_kernel(const int* __restrict__ idx, const float* __restrict__ emb,
                             bf16_t* __restrict__ out, int B_, int L_, int C_){
  int t = blockIdx.x * blockDim.x + threadIdx.x;
  int total = B_ * C_ * L_;
  if (t >= total) return;
  int b = t / (C_ * L_);
  int r = t % (C_ * L_);
  int c = r / L_;
  int l = r % L_;
  int id = idx[b * L_ + l];
  out[t] = (bf16_t)emb[id * C_ + c];
}

// ------------------------------ WMMA GEMM ----------------------------------
// C[M,N] = act(A[M,K] * B[K,N] + bias[M])
// BMODE:  B_IM2COL -> B[k][n] = X[b][ci][l+kw]   (k=ci*KW+kw, n=b*Lout+l)
//         B_TRANS  -> B[k][n] = Bsrc[n*K + k]    (row-major [N,K] source)
//         B_DENSE  -> B[k][n] = Bsrc[k*N + n]    (row-major [K,N] source)
// K is always a multiple of 32; for B_DENSE, N is a multiple of 8.

enum { B_IM2COL = 0, B_TRANS = 1, B_DENSE = 2 };
enum { OUT_XLAYOUT = 0, OUT_XC = 1, OUT_MN_BF16 = 2, OUT_MN_F32 = 3, OUT_NM_BF16 = 4 };
enum { ACT_NONE = 0, ACT_RELU = 1, ACT_LEAKY = 2 };

#define BP_STRIDE 24   // elems per permuted-B row (16 data + 8 pad, keeps 16B align)

template<int BMODE, int OUTMODE, int ACT>
__global__ __launch_bounds__(128) void wmma_gemm(
    const bf16_t* __restrict__ A, const float* __restrict__ bias,
    const bf16_t* __restrict__ Bsrc,
    int M, int N, int K,
    int Cin, int KW, int Lin, int Lout,
    void* __restrict__ Cout)
{
  // A tile: [m][k] row-major (rows of 32 bf16, 40-elem stride = 80B, 16B-aligned chunks)
  __shared__ __align__(16) bf16_t As[64][40];
  // B tile: WMMA-fragment-permuted: row = (nt*2 + khalf)*16 + nl, 16 elems + pad
  __shared__ __align__(16) bf16_t Bp[128 * BP_STRIDE];

  int tid  = threadIdx.x;
  int lane = tid & 31;
  int wave = tid >> 5;
  int half = lane >> 4;
  int lm   = lane & 15;
  int nBase = blockIdx.x * 64;
  int mBase = blockIdx.y * 64;

  v8f acc[4];
  #pragma unroll
  for (int nt = 0; nt < 4; nt++)
    #pragma unroll
    for (int i = 0; i < 8; i++) acc[nt][i] = 0.0f;

  for (int kBase = 0; kBase < K; kBase += 32){
    // ---- A tile: 64 rows x 32 cols as 8-elem (16B) vector loads ----
    for (int ch = tid; ch < 256; ch += 128){
      int r  = ch >> 2;
      int cc = (ch & 3) * 8;
      int gm = mBase + r;
      uint4 val = make_uint4(0u, 0u, 0u, 0u);
      if (gm < M) val = *(const uint4*)(A + (size_t)gm * K + (kBase + cc));
      *(uint4*)&As[r][cc] = val;
    }
    // ---- B tile into WMMA-permuted LDS ----
    if (BMODE == B_TRANS){
      for (int ch = tid; ch < 256; ch += 128){
        int n  = ch >> 2;      // n local 0..63
        int kc = ch & 3;       // 8-elem k chunk 0..3
        int gn = nBase + n;
        uint4 val = make_uint4(0u, 0u, 0u, 0u);
        if (gn < N) val = *(const uint4*)(Bsrc + (size_t)gn * K + (kBase + kc * 8));
        int nt = n >> 4, nl = n & 15;
        int khalf = kc >> 1, klo = (kc & 1) * 8;
        *(uint4*)&Bp[(((nt * 2 + khalf) * 16) + nl) * BP_STRIDE + klo] = val;
      }
    } else if (BMODE == B_DENSE){
      for (int ch = tid; ch < 256; ch += 128){
        int k  = ch >> 3;      // k local 0..31
        int nc = ch & 7;       // 8-elem n chunk 0..7
        int gk = kBase + k;
        int gn = nBase + nc * 8;
        Pack8 val; val.q = make_uint4(0u, 0u, 0u, 0u);
        if (gn < N) val.q = *(const uint4*)(Bsrc + (size_t)gk * N + gn);
        int khalf = k >> 4, ki = k & 15;
        #pragma unroll
        for (int j = 0; j < 8; j++){
          int n  = nc * 8 + j;
          int nt = n >> 4, nl = n & 15;
          Bp[(((nt * 2 + khalf) * 16) + nl) * BP_STRIDE + ki] = val.e[j];
        }
      }
    } else {
      for (int i = tid; i < 2048; i += 128){
        int k = i >> 6, n = i & 63;
        int gk = kBase + k, gn = nBase + n;
        bf16_t v = (bf16_t)0.0f;
        if (gn < N){
          int ci = gk / KW, kw = gk % KW;
          int bb = gn / Lout, l = gn % Lout;
          v = Bsrc[((size_t)bb * Cin + ci) * Lin + l + kw];
        }
        int nt = n >> 4, nl = n & 15;
        int khalf = k >> 4, ki = k & 15;
        Bp[(((nt * 2 + khalf) * 16) + nl) * BP_STRIDE + ki] = v;
      }
    }
    __syncthreads();

    // A fragment: two b128 chunks, contiguous k = half*16 + s (matches Bp)
    FragU fa;
    {
      const bf16_t* arow = &As[wave * 16 + lm][0];
      fa.q[0] = *(const uint4*)(arow + half * 16);
      fa.q[1] = *(const uint4*)(arow + half * 16 + 8);
    }
    #pragma unroll
    for (int nt = 0; nt < 4; nt++){
      FragU fb;
      const bf16_t* brow = &Bp[(((nt * 2 + half) * 16) + lm) * BP_STRIDE];
      fb.q[0] = *(const uint4*)(brow);
      fb.q[1] = *(const uint4*)(brow + 8);
      acc[nt] = __builtin_amdgcn_wmma_f32_16x16x32_bf16(
          false, fa.v, false, fb.v, (short)0, acc[nt], false, false);
    }
    __syncthreads();
  }

  // epilogue: D layout -> m_local = half*8 + v, n_local = lm
  #pragma unroll
  for (int nt = 0; nt < 4; nt++){
    #pragma unroll
    for (int v = 0; v < 8; v++){
      int m = mBase + wave * 16 + half * 8 + v;
      int n = nBase + nt * 16 + lm;
      if (m < M && n < N){
        float val = acc[nt][v] + bias[m];
        if (ACT == ACT_RELU)  val = fmaxf(val, 0.0f);
        if (ACT == ACT_LEAKY) val = (val > 0.0f) ? val : 0.01f * val;
        if (OUTMODE == OUT_XLAYOUT){            // [B][Cout][Lout] bf16
          int bb = n / Lout, l = n % Lout;
          ((bf16_t*)Cout)[((size_t)bb * M + m) * Lout + l] = (bf16_t)val;
        } else if (OUTMODE == OUT_XC){          // [(l*32+b)][256] bf16
          int bb = n / Lout, l = n % Lout;
          ((bf16_t*)Cout)[(size_t)(l * 32 + bb) * M + m] = (bf16_t)val;
        } else if (OUTMODE == OUT_MN_BF16){     // [M][N] bf16
          ((bf16_t*)Cout)[(size_t)m * N + n] = (bf16_t)val;
        } else if (OUTMODE == OUT_MN_F32){      // [M][N] f32
          ((float*)Cout)[(size_t)m * N + n] = val;
        } else {                                // [N][M] bf16
          ((bf16_t*)Cout)[(size_t)n * M + m] = (bf16_t)val;
        }
      }
    }
  }
}

// ---------------------------- CBAM-style block -----------------------------

__global__ __launch_bounds__(256) void rowstat_kernel(const bf16_t* __restrict__ Xc,
                                                      float* __restrict__ mx,
                                                      float* __restrict__ av){
  int l = blockIdx.x, tid = threadIdx.x;
  const bf16_t* basep = Xc + (size_t)l * 32 * 256;
  float m = -3.0e38f, s = 0.0f;
  for (int i = tid; i < 32 * 256; i += 256){
    float v = (float)basep[i];
    m = fmaxf(m, v); s += v;
  }
  __shared__ float smx[256], ssm[256];
  smx[tid] = m; ssm[tid] = s;
  __syncthreads();
  for (int st = 128; st > 0; st >>= 1){
    if (tid < st){ smx[tid] = fmaxf(smx[tid], smx[tid + st]); ssm[tid] += ssm[tid + st]; }
    __syncthreads();
  }
  if (tid == 0){ mx[l] = smx[0]; av[l] = ssm[0] / 8192.0f; }
}

__global__ __launch_bounds__(256) void se_cm_kernel(const float* __restrict__ se1,
                                                    const float* __restrict__ se2,
                                                    const float* __restrict__ mx,
                                                    const float* __restrict__ av,
                                                    float* __restrict__ cm, int L, int Hd){
  __shared__ float hsum[64];
  int tid = threadIdx.x;
  for (int j = tid; j < Hd; j += 256){
    float s1 = 0.f, s2 = 0.f;
    for (int i = 0; i < L; i++){ float w = se1[j * L + i]; s1 += w * mx[i]; s2 += w * av[i]; }
    hsum[j] = fmaxf(s1, 0.f) + fmaxf(s2, 0.f);
  }
  __syncthreads();
  for (int l = tid; l < L; l += 256){
    float s = 0.f;
    for (int j = 0; j < Hd; j++) s += se2[l * Hd + j] * hsum[j];
    cm[l] = sigmoidf_(s);
  }
}

__global__ void spat_stat_kernel(const bf16_t* __restrict__ Xc, float* __restrict__ sst, int L){
  int t = blockIdx.x * blockDim.x + threadIdx.x;
  if (t >= 32 * 256) return;
  int b = t >> 8, c = t & 255;
  float m = -3.0e38f, s = 0.f;
  for (int l = 0; l < L; l++){
    float v = (float)Xc[((size_t)l * 32 + b) * 256 + c];
    m = fmaxf(m, v); s += v;
  }
  sst[t] = m;
  sst[8192 + t] = s / (float)L;
}

__global__ void spat_conv_kernel(const float* __restrict__ sst, const float* __restrict__ sw,
                                 const float* __restrict__ sb, float* __restrict__ smo){
  int t = blockIdx.x * blockDim.x + threadIdx.x;
  if (t >= 32 * 256) return;
  int b = t >> 8, c = t & 255;
  float acc = sb[0];
  for (int i = 0; i < 2; i++)
    for (int dy = 0; dy < 9; dy++){
      int bb = b + dy - 4;
      if (bb < 0 || bb >= 32) continue;
      for (int dx = 0; dx < 9; dx++){
        int cc = c + dx - 4;
        if (cc < 0 || cc >= 256) continue;
        acc += sst[i * 8192 + bb * 256 + cc] * sw[(i * 9 + dy) * 9 + dx];
      }
    }
  smo[t] = sigmoidf_(acc);
}

__global__ void amap_kernel(const float* __restrict__ cm, const float* __restrict__ smo,
                            bf16_t* __restrict__ amap, int R){
  int t = blockIdx.x * blockDim.x + threadIdx.x;
  if (t >= R * 256) return;
  int r = t >> 8, c = t & 255;
  int l = r >> 5, b = r & 31;
  amap[t] = (bf16_t)(cm[l] * smo[b * 256 + c]);
}

// ---------------------------- attention core -------------------------------
// One wave per (l, head). dh == seq == 32 == WMMA bf16 K.
// Q,K in [R,256] bf16; V, Aproj in [256,R] bf16; O out [256,R] bf16.
// Every fragment / epilogue access is a b128 (or pair) vector op.
__global__ __launch_bounds__(32) void attn_kernel(
    const bf16_t* __restrict__ Q, const bf16_t* __restrict__ Km,
    const bf16_t* __restrict__ V, const bf16_t* __restrict__ Ap,
    bf16_t* __restrict__ O, int L, int R)
{
  int l = blockIdx.x >> 3;
  int h = blockIdx.x & 7;
  int lane = threadIdx.x;
  int half = lane >> 4, lm = lane & 15;

  __shared__ float en[32][33];
  __shared__ __align__(16) bf16_t pr[32][32];   // 64B row stride, 16B-aligned chunks
  const float scale = 0.17677669529663689f;     // 1/sqrt(32)

  // energy = (Q Kt)/sqrt(dh) + A
  #pragma unroll
  for (int qt = 0; qt < 2; qt++){
    FragU fa;
    {
      const bf16_t* qrow = Q + (size_t)(l * 32 + qt * 16 + lm) * 256 + h * 32;
      fa.q[0] = *(const uint4*)(qrow + half * 16);
      fa.q[1] = *(const uint4*)(qrow + half * 16 + 8);
    }
    #pragma unroll
    for (int kt = 0; kt < 2; kt++){
      FragU fb;
      {
        const bf16_t* krow = Km + (size_t)(l * 32 + kt * 16 + lm) * 256 + h * 32;
        fb.q[0] = *(const uint4*)(krow + half * 16);
        fb.q[1] = *(const uint4*)(krow + half * 16 + 8);
      }
      v8f c;
      #pragma unroll
      for (int i = 0; i < 8; i++) c[i] = 0.0f;
      c = __builtin_amdgcn_wmma_f32_16x16x32_bf16(false, fa.v, false, fb.v, (short)0, c, false, false);
      // A-bias tile: lane's 8 accumulator rows are contiguous in [256,R] layout
      Pack8 ap;
      ap.q = *(const uint4*)(Ap + (size_t)(h * 32 + kt * 16 + lm) * R
                                + l * 32 + qt * 16 + half * 8);
      #pragma unroll
      for (int v = 0; v < 8; v++){
        int q  = qt * 16 + half * 8 + v;
        int k2 = kt * 16 + lm;
        en[q][k2] = c[v] * scale + (float)ap.e[v];
      }
    }
  }
  __syncthreads();

  // row softmax: one row of 32 per lane
  {
    int q = lane;
    float m = en[q][0];
    #pragma unroll
    for (int k2 = 1; k2 < 32; k2++) m = fmaxf(m, en[q][k2]);
    float s = 0.f, ex[32];
    #pragma unroll
    for (int k2 = 0; k2 < 32; k2++){ ex[k2] = __expf(en[q][k2] - m); s += ex[k2]; }
    float inv = 1.f / s;
    #pragma unroll
    for (int k2 = 0; k2 < 32; k2++) pr[q][k2] = (bf16_t)(ex[k2] * inv);
  }
  __syncthreads();

  // O = P Vblk^T, stored [256,R] so each lane's 8 outputs pack into one b128
  #pragma unroll
  for (int qt = 0; qt < 2; qt++){
    FragU fa;
    {
      const bf16_t* prow = &pr[qt * 16 + lm][0];
      fa.q[0] = *(const uint4*)(prow + half * 16);
      fa.q[1] = *(const uint4*)(prow + half * 16 + 8);
    }
    #pragma unroll
    for (int dt = 0; dt < 2; dt++){
      FragU fb;
      {
        const bf16_t* vrow = V + (size_t)(h * 32 + dt * 16 + lm) * R + l * 32;
        fb.q[0] = *(const uint4*)(vrow + half * 16);
        fb.q[1] = *(const uint4*)(vrow + half * 16 + 8);
      }
      v8f c;
      #pragma unroll
      for (int i = 0; i < 8; i++) c[i] = 0.0f;
      c = __builtin_amdgcn_wmma_f32_16x16x32_bf16(false, fa.v, false, fb.v, (short)0, c, false, false);
      Pack8 ov;
      #pragma unroll
      for (int v = 0; v < 8; v++) ov.e[v] = (bf16_t)c[v];
      *(uint4*)(O + (size_t)(h * 32 + dt * 16 + lm) * R
                  + l * 32 + qt * 16 + half * 8) = ov.q;
    }
  }
}

// 0.5*x + 0.5*mhsa, then max over L; concat drug|prot -> pair[32][512] bf16
__global__ void combine_kernel(const bf16_t* __restrict__ XcD, const float* __restrict__ WFD,
                               int LD_, int RD,
                               const bf16_t* __restrict__ XcP, const float* __restrict__ WFP,
                               int LP_, int RP,
                               bf16_t* __restrict__ pair){
  int t = blockIdx.x * blockDim.x + threadIdx.x;
  if (t >= 32 * 512) return;
  int b = t >> 9, c = t & 511;
  float best = -3.0e38f;
  if (c < 256){
    for (int l = 0; l < LD_; l++){
      float v = 0.5f * (float)XcD[((size_t)l * 32 + b) * 256 + c]
              + 0.5f * WFD[(size_t)c * RD + l * 32 + b];
      best = fmaxf(best, v);
    }
  } else {
    int c2 = c - 256;
    for (int l = 0; l < LP_; l++){
      float v = 0.5f * (float)XcP[((size_t)l * 32 + b) * 256 + c2]
              + 0.5f * WFP[(size_t)c2 * RP + l * 32 + b];
      best = fmaxf(best, v);
    }
  }
  pair[t] = (bf16_t)best;
}

__global__ void out_kernel(const bf16_t* __restrict__ h3, const float* __restrict__ w,
                           const float* __restrict__ bo, float* __restrict__ out){
  int t = blockIdx.x * blockDim.x + threadIdx.x;
  if (t >= 64) return;
  int b = t >> 1, j = t & 1;
  float s = bo[j];
  for (int k = 0; k < 512; k++) s += (float)h3[b * 512 + k] * w[j * 512 + k];
  out[b * 2 + j] = s;
}

// ------------------------------- host side ---------------------------------

static void run_branch(
    const float* embT, int idxL, const int* idx,
    const bf16_t* w1, const float* b1, int k1,
    const bf16_t* w2, const float* b2, int k2,
    const bf16_t* w3, const float* b3, int k3,
    const float* se1, const float* se2, int Hd,
    const float* sw, const float* sb,
    const bf16_t* wq, const bf16_t* wk, const bf16_t* wv,
    const bf16_t* wf, const bf16_t* wa,
    const float* bq, const float* bk, const float* bv,
    const float* bfb, const float* ba,
    bf16_t* EMB, bf16_t* C1, bf16_t* C2,
    bf16_t* XC, bf16_t* Qb, bf16_t* Kb, bf16_t* Vb, bf16_t* Ab,
    bf16_t* AM, bf16_t* Ob, float* WF,
    float* mx, float* av, float* cm, float* sst, float* smo,
    hipStream_t stream)
{
  int L1 = idxL - k1 + 1;
  int L2 = L1 - k2 + 1;
  int L3 = L2 - k3 + 1;
  int R  = L3 * 32;

  { int total = 32 * 64 * idxL;
    embed_kernel<<<(total + 255) / 256, 256, 0, stream>>>(idx, embT, EMB, 32, idxL, 64); }

  { int M = 64, K = 64 * k1, N = 32 * L1;
    dim3 g((N + 63) / 64, (M + 63) / 64);
    wmma_gemm<B_IM2COL, OUT_XLAYOUT, ACT_RELU><<<g, 128, 0, stream>>>(
        w1, b1, EMB, M, N, K, 64, k1, idxL, L1, C1); }

  { int M = 128, K = 64 * k2, N = 32 * L2;
    dim3 g((N + 63) / 64, (M + 63) / 64);
    wmma_gemm<B_IM2COL, OUT_XLAYOUT, ACT_RELU><<<g, 128, 0, stream>>>(
        w2, b2, C1, M, N, K, 64, k2, L1, L2, C2); }

  { int M = 256, K = 128 * k3, N = 32 * L3;
    dim3 g((N + 63) / 64, (M + 63) / 64);
    wmma_gemm<B_IM2COL, OUT_XC, ACT_RELU><<<g, 128, 0, stream>>>(
        w3, b3, C2, M, N, K, 128, k3, L2, L3, XC); }

  rowstat_kernel<<<L3, 256, 0, stream>>>(XC, mx, av);
  se_cm_kernel<<<1, 256, 0, stream>>>(se1, se2, mx, av, cm, L3, Hd);
  spat_stat_kernel<<<32, 256, 0, stream>>>(XC, sst, L3);
  spat_conv_kernel<<<32, 256, 0, stream>>>(sst, sw, sb, smo);
  { int total = R * 256;
    amap_kernel<<<(total + 255) / 256, 256, 0, stream>>>(cm, smo, AM, R); }

  { int M = 256, N = R, K = 256;
    dim3 g((N + 63) / 64, (M + 63) / 64);
    wmma_gemm<B_TRANS, OUT_NM_BF16, ACT_NONE><<<g, 128, 0, stream>>>(
        wq, bq, XC, M, N, K, 0, 1, 0, 1, Qb);
    wmma_gemm<B_TRANS, OUT_NM_BF16, ACT_NONE><<<g, 128, 0, stream>>>(
        wk, bk, XC, M, N, K, 0, 1, 0, 1, Kb);
    wmma_gemm<B_TRANS, OUT_MN_BF16, ACT_NONE><<<g, 128, 0, stream>>>(
        wv, bv, XC, M, N, K, 0, 1, 0, 1, Vb);
    wmma_gemm<B_TRANS, OUT_MN_BF16, ACT_NONE><<<g, 128, 0, stream>>>(
        wa, ba, AM, M, N, K, 0, 1, 0, 1, Ab); }

  attn_kernel<<<L3 * 8, 32, 0, stream>>>(Qb, Kb, Vb, Ab, Ob, L3, R);

  // wf projection: B = O in row-major [K=256, N=R]
  { int M = 256, N = R, K = 256;
    dim3 g((N + 63) / 64, (M + 63) / 64);
    wmma_gemm<B_DENSE, OUT_MN_F32, ACT_NONE><<<g, 128, 0, stream>>>(
        wf, bfb, Ob, M, N, K, 0, 1, 0, 1, WF); }
}

extern "C" void kernel_launch(void* const* d_in, const int* in_sizes, int n_in,
                              void* d_out, int out_size, void* d_ws, size_t ws_size,
                              hipStream_t stream) {
  (void)in_sizes; (void)n_in; (void)out_size; (void)ws_size;

  // ---- inputs (setup_inputs dict order, leaves flattened) ----
  const float* drug_emb = (const float*)d_in[0];
  const float* prot_emb = (const float*)d_in[1];
  const float* d_w1 = (const float*)d_in[2];  const float* d_b1 = (const float*)d_in[3];
  const float* d_w2 = (const float*)d_in[4];  const float* d_b2 = (const float*)d_in[5];
  const float* d_w3 = (const float*)d_in[6];  const float* d_b3 = (const float*)d_in[7];
  const float* p_w1 = (const float*)d_in[8];  const float* p_b1 = (const float*)d_in[9];
  const float* p_w2 = (const float*)d_in[10]; const float* p_b2 = (const float*)d_in[11];
  const float* p_w3 = (const float*)d_in[12]; const float* p_b3 = (const float*)d_in[13];
  const float* cbd_se1 = (const float*)d_in[14]; const float* cbd_se2 = (const float*)d_in[15];
  const float* cbd_sw  = (const float*)d_in[16]; const float* cbd_sb  = (const float*)d_in[17];
  const float* cbp_se1 = (const float*)d_in[18]; const float* cbp_se2 = (const float*)d_in[19];
  const float* cbp_sw  = (const float*)d_in[20]; const float* cbp_sb  = (const float*)d_in[21];
  const float* wq_f = (const float*)d_in[22]; const float* wk_f = (const float*)d_in[23];
  const float* wv_f = (const float*)d_in[24]; const float* wf_f = (const float*)d_in[25];
  const float* wa_f = (const float*)d_in[26];
  const float* bq = (const float*)d_in[27]; const float* bk = (const float*)d_in[28];
  const float* bv = (const float*)d_in[29]; const float* bfb = (const float*)d_in[30];
  const float* ba = (const float*)d_in[31];
  const float* f1w = (const float*)d_in[32]; const float* f1b = (const float*)d_in[33];
  const float* f2w = (const float*)d_in[34]; const float* f2b = (const float*)d_in[35];
  const float* f3w = (const float*)d_in[36]; const float* f3b = (const float*)d_in[37];
  const float* ow  = (const float*)d_in[38]; const float* ob  = (const float*)d_in[39];
  const int* drug = (const int*)d_in[40];
  const int* prot = (const int*)d_in[41];

  // ---- workspace layout (bump allocator) ----
  char* base = (char*)d_ws;
  size_t off = 0;
  auto alloc = [&](size_t nbytes) -> void* {
    void* r = (void*)(base + off);
    off += (nbytes + 255) & ~(size_t)255;
    return r;
  };

  const int RD = 85 * 32;        // 2720
  const int RP = 979 * 32;       // 31328
  const size_t RMAX = (size_t)RP;

  // bf16 weight copies
  bf16_t* wd1 = (bf16_t*)alloc((size_t)64 * 64 * 4 * 2);
  bf16_t* wd2 = (bf16_t*)alloc((size_t)128 * 64 * 6 * 2);
  bf16_t* wd3 = (bf16_t*)alloc((size_t)256 * 128 * 8 * 2);
  bf16_t* wp1 = (bf16_t*)alloc((size_t)64 * 64 * 4 * 2);
  bf16_t* wp2 = (bf16_t*)alloc((size_t)128 * 64 * 8 * 2);
  bf16_t* wp3 = (bf16_t*)alloc((size_t)256 * 128 * 12 * 2);
  bf16_t* wqb = (bf16_t*)alloc((size_t)256 * 256 * 2);
  bf16_t* wkb = (bf16_t*)alloc((size_t)256 * 256 * 2);
  bf16_t* wvb = (bf16_t*)alloc((size_t)256 * 256 * 2);
  bf16_t* wfbb= (bf16_t*)alloc((size_t)256 * 256 * 2);
  bf16_t* wab = (bf16_t*)alloc((size_t)256 * 256 * 2);
  bf16_t* f1wb= (bf16_t*)alloc((size_t)1024 * 512 * 2);
  bf16_t* f2wb= (bf16_t*)alloc((size_t)1024 * 1024 * 2);
  bf16_t* f3wb= (bf16_t*)alloc((size_t)512 * 1024 * 2);

  // shared activation scratch (sized for protein branch)
  bf16_t* EMB = (bf16_t*)alloc((size_t)32 * 64 * 1000 * 2);
  bf16_t* C1  = (bf16_t*)alloc((size_t)32 * 64 * 997 * 2);
  bf16_t* C2  = (bf16_t*)alloc((size_t)32 * 128 * 990 * 2);
  bf16_t* Qb  = (bf16_t*)alloc(RMAX * 256 * 2);
  bf16_t* Kb  = (bf16_t*)alloc(RMAX * 256 * 2);
  bf16_t* Vb  = (bf16_t*)alloc(RMAX * 256 * 2);
  bf16_t* Ab  = (bf16_t*)alloc(RMAX * 256 * 2);
  bf16_t* AM  = (bf16_t*)alloc(RMAX * 256 * 2);
  bf16_t* Ob  = (bf16_t*)alloc(RMAX * 256 * 2);
  // per-branch persistent results
  bf16_t* XCD = (bf16_t*)alloc((size_t)RD * 256 * 2);
  bf16_t* XCP = (bf16_t*)alloc((size_t)RP * 256 * 2);
  float*  WFD = (float*)alloc((size_t)RD * 256 * 4);
  float*  WFP = (float*)alloc((size_t)RP * 256 * 4);
  // small buffers
  float* mx  = (float*)alloc(979 * 4);
  float* av  = (float*)alloc(979 * 4);
  float* cm  = (float*)alloc(979 * 4);
  float* sst = (float*)alloc(2 * 8192 * 4);
  float* smo = (float*)alloc(8192 * 4);
  bf16_t* pairB = (bf16_t*)alloc(32 * 512 * 2);
  bf16_t* h1 = (bf16_t*)alloc(32 * 1024 * 2);
  bf16_t* h2 = (bf16_t*)alloc(32 * 1024 * 2);
  bf16_t* h3 = (bf16_t*)alloc(32 * 512 * 2);

  // ---- convert weights to bf16 ----
  auto cvt = [&](const float* s, bf16_t* d, int n){
    cvt_f32_bf16<<<(n + 255) / 256, 256, 0, stream>>>(s, d, n);
  };
  cvt(d_w1, wd1, 64 * 64 * 4);   cvt(d_w2, wd2, 128 * 64 * 6);  cvt(d_w3, wd3, 256 * 128 * 8);
  cvt(p_w1, wp1, 64 * 64 * 4);   cvt(p_w2, wp2, 128 * 64 * 8);  cvt(p_w3, wp3, 256 * 128 * 12);
  cvt(wq_f, wqb, 65536); cvt(wk_f, wkb, 65536); cvt(wv_f, wvb, 65536);
  cvt(wf_f, wfbb, 65536); cvt(wa_f, wab, 65536);
  cvt(f1w, f1wb, 1024 * 512); cvt(f2w, f2wb, 1024 * 1024); cvt(f3w, f3wb, 512 * 1024);

  // ---- drug branch (L: 100 ->97 ->92 ->85) ----
  run_branch(drug_emb, 100, drug,
             wd1, d_b1, 4, wd2, d_b2, 6, wd3, d_b3, 8,
             cbd_se1, cbd_se2, 5, cbd_sw, cbd_sb,
             wqb, wkb, wvb, wfbb, wab, bq, bk, bv, bfb, ba,
             EMB, C1, C2, XCD, Qb, Kb, Vb, Ab, AM, Ob, WFD,
             mx, av, cm, sst, smo, stream);

  // ---- protein branch (L: 1000 ->997 ->990 ->979) ----
  run_branch(prot_emb, 1000, prot,
             wp1, p_b1, 4, wp2, p_b2, 8, wp3, p_b3, 12,
             cbp_se1, cbp_se2, 61, cbp_sw, cbp_sb,
             wqb, wkb, wvb, wfbb, wab, bq, bk, bv, bfb, ba,
             EMB, C1, C2, XCP, Qb, Kb, Vb, Ab, AM, Ob, WFP,
             mx, av, cm, sst, smo, stream);

  // ---- combine + max-pool over L, concat -> pair[32][512] ----
  combine_kernel<<<(32 * 512 + 255) / 256, 256, 0, stream>>>(
      XCD, WFD, 85, RD, XCP, WFP, 979, RP, pairB);

  // ---- FC stack (WMMA GEMMs, leaky relu) ----
  { int M = 1024, N = 32, K = 512;
    dim3 g((N + 63) / 64, (M + 63) / 64);
    wmma_gemm<B_TRANS, OUT_NM_BF16, ACT_LEAKY><<<g, 128, 0, stream>>>(
        f1wb, f1b, pairB, M, N, K, 0, 1, 0, 1, h1); }
  { int M = 1024, N = 32, K = 1024;
    dim3 g((N + 63) / 64, (M + 63) / 64);
    wmma_gemm<B_TRANS, OUT_NM_BF16, ACT_LEAKY><<<g, 128, 0, stream>>>(
        f2wb, f2b, h1, M, N, K, 0, 1, 0, 1, h2); }
  { int M = 512, N = 32, K = 1024;
    dim3 g((N + 63) / 64, (M + 63) / 64);
    wmma_gemm<B_TRANS, OUT_NM_BF16, ACT_LEAKY><<<g, 128, 0, stream>>>(
        f3wb, f3b, h2, M, N, K, 0, 1, 0, 1, h3); }

  out_kernel<<<1, 64, 0, stream>>>(h3, ow, ob, (float*)d_out);
}
